// HyperAttention_41738492182539
// MI455X (gfx1250) — compile-verified, matrix operand order checked by reading, not run on hardware
//
#include <hip/hip_runtime.h>

#define NN 8192
#define DD 64
#define BH 32          // B*H
#define NBLK 32        // number of 256-row blocks
#define BS 256
#define SS 256
#define NPROJ 7
#define SCALEF 0.125f
#define MASK_NEG (-3.0e38f)
#define M_INIT (-1.0e30f)
#define LOG32 3.4657359027997265f

typedef __attribute__((ext_vector_type(16))) __bf16 v16bf;
typedef __attribute__((ext_vector_type(8)))  float  v8f;

union FragU { uint4 u[2]; v16bf v; };

__device__ __forceinline__ unsigned short f2bf(float x) {
  unsigned int u = __float_as_uint(x);
  u += 0x7fffu + ((u >> 16) & 1u);
  return (unsigned short)(u >> 16);
}

// ---------------- Kernel 1: LSH hash (gray(bin) = bin ^ (bin>>1)) -------------
__global__ void hash_kernel(const float* __restrict__ q, const float* __restrict__ k,
                            const float* __restrict__ proj,
                            unsigned int* __restrict__ qh, unsigned int* __restrict__ kh) {
  const int w = threadIdx.x >> 5, lane = threadIdx.x & 31;
  long long row = (long long)blockIdx.x * 8 + w;           // [0, 2*BH*NN)
  const long long TOT = (long long)BH * NN;
  const float* src;
  unsigned int* dst;
  if (row < TOT) { src = q + row * DD; dst = qh + row; }
  else           { long long r2 = row - TOT; src = k + r2 * DD; dst = kh + r2; }
  float x0 = src[lane], x1 = src[lane + 32];
  unsigned int bits = 0;
#pragma unroll
  for (int p = 0; p < NPROJ; ++p) {
    float a = x0 * proj[lane * NPROJ + p] + x1 * proj[(lane + 32) * NPROJ + p];
    a += __shfl_xor(a, 1, 32);  a += __shfl_xor(a, 2, 32);
    a += __shfl_xor(a, 4, 32);  a += __shfl_xor(a, 8, 32);
    a += __shfl_xor(a, 16, 32);
    bits |= (a > 0.0f ? 1u : 0u) << p;
  }
  unsigned int h = bits ^ (bits >> 1);
  if (lane == 0) *dst = h;
}

// ------------- Kernel 2: stable counting sort over 128 buckets ----------------
__global__ void __launch_bounds__(64)
sort_kernel(const unsigned int* __restrict__ qh, const unsigned int* __restrict__ kh,
            int* __restrict__ qsi, int* __restrict__ ksi) {
  __shared__ unsigned int hist[128 * 64];   // [bucket][thread], bucket-major
  __shared__ unsigned int tot[64];
  const int t = threadIdx.x;
  const bool isq = blockIdx.x < BH;
  const int bh = isq ? blockIdx.x : (blockIdx.x - BH);
  const unsigned int* h = (isq ? qh : kh) + (long long)bh * NN;
  int* si = (isq ? qsi : ksi) + (long long)bh * NN;

  for (int i = t; i < 128 * 64; i += 64) hist[i] = 0;
  __syncthreads();
  const int base = t * 128;                 // contiguous chunk per thread
  for (int j = 0; j < 128; ++j) {
    unsigned int b = h[base + j];
    hist[b * 64 + t]++;                     // column t is private
  }
  __syncthreads();
  // exclusive scan over flattened (bucket*64 + thread)
  unsigned int s = 0;
  for (int f = 0; f < 128; ++f) s += hist[t * 128 + f];
  tot[t] = s;
  __syncthreads();
  if (t == 0) {
    unsigned int run = 0;
    for (int i = 0; i < 64; ++i) { unsigned int tmp = tot[i]; tot[i] = run; run += tmp; }
  }
  __syncthreads();
  unsigned int val = tot[t];
  for (int f = 0; f < 128; ++f) { unsigned int tmp = hist[t * 128 + f]; hist[t * 128 + f] = val; val += tmp; }
  __syncthreads();
  for (int j = 0; j < 128; ++j) {
    int i = base + j;
    unsigned int b = h[i];
    unsigned int p = hist[b * 64 + t]++;
    si[p] = i;                              // sorted position p holds original row i
  }
}

// -------- Kernel 3: gather sorted Q,K (bf16) and V transposed [d][n] ----------
__global__ void gather_qkv(const float* __restrict__ q, const float* __restrict__ k,
                           const float* __restrict__ v,
                           const int* __restrict__ qsi, const int* __restrict__ ksi,
                           unsigned short* __restrict__ qs, unsigned short* __restrict__ ks,
                           unsigned short* __restrict__ vst) {
  long long t = (long long)blockIdx.x * blockDim.x + threadIdx.x;  // BH*NN*4 threads
  long long row = t >> 2; int c = (int)(t & 3);
  int bh = (int)(row >> 13); int j = (int)(row & 8191);
  long long bo = (long long)bh * NN;
  {
    const float* s = q + (bo + qsi[bo + j]) * DD + c * 16;
    unsigned int pk[8];
#pragma unroll
    for (int i = 0; i < 8; ++i) pk[i] = (unsigned int)f2bf(s[2 * i]) | ((unsigned int)f2bf(s[2 * i + 1]) << 16);
    uint4* d = (uint4*)(qs + row * DD + c * 16);
    d[0] = make_uint4(pk[0], pk[1], pk[2], pk[3]);
    d[1] = make_uint4(pk[4], pk[5], pk[6], pk[7]);
  }
  {
    int kr = ksi[bo + j];
    const float* s = k + (bo + kr) * DD + c * 16;
    unsigned int pk[8];
#pragma unroll
    for (int i = 0; i < 8; ++i) pk[i] = (unsigned int)f2bf(s[2 * i]) | ((unsigned int)f2bf(s[2 * i + 1]) << 16);
    uint4* d = (uint4*)(ks + row * DD + c * 16);
    d[0] = make_uint4(pk[0], pk[1], pk[2], pk[3]);
    d[1] = make_uint4(pk[4], pk[5], pk[6], pk[7]);
    const float* sv = v + (bo + kr) * DD + c * 16;
    unsigned short* dv = vst + (long long)bh * DD * NN + (long long)(c * 16) * NN + j;
#pragma unroll
    for (int i = 0; i < 16; ++i) dv[(long long)i * NN] = f2bf(sv[i]);
  }
}

// ------ Kernel 4: gather sampled keys (bf16), Vsub transposed, block ids ------
__global__ void gather_sub(const float* __restrict__ k, const float* __restrict__ v,
                           const int* __restrict__ ksi, const int* __restrict__ sampled,
                           unsigned short* __restrict__ ksub, unsigned short* __restrict__ vsubt,
                           int* __restrict__ sblk) {
  int t = blockIdx.x * blockDim.x + threadIdx.x;   // BH*SS*4 threads
  int row = t >> 2; int c = t & 3;
  int bh = row >> 8; int s = row & 255;
  int j = sampled[bh * SS + s];                    // sorted-order position
  long long bo = (long long)bh * NN;
  int kr = ksi[bo + j];
  {
    const float* sp = k + (bo + kr) * DD + c * 16;
    unsigned int pk[8];
#pragma unroll
    for (int i = 0; i < 8; ++i) pk[i] = (unsigned int)f2bf(sp[2 * i]) | ((unsigned int)f2bf(sp[2 * i + 1]) << 16);
    uint4* d = (uint4*)(ksub + (long long)row * DD + c * 16);
    d[0] = make_uint4(pk[0], pk[1], pk[2], pk[3]);
    d[1] = make_uint4(pk[4], pk[5], pk[6], pk[7]);
  }
  {
    const float* vp = v + (bo + kr) * DD + c * 16;
    unsigned short* dv = vsubt + (long long)bh * DD * SS + (long long)(c * 16) * SS + s;
#pragma unroll
    for (int i = 0; i < 16; ++i) dv[i * SS] = f2bf(vp[i]);
  }
  if (c == 0) sblk[bh * SS + s] = j >> 8;          // sampled block id
}

// --------- Kernel 5/6: flash attention over 256 keys with bf16 WMMA -----------
template <int RES>
__global__ void __launch_bounds__(256)
attn_kernel(const unsigned short* __restrict__ qs,
            const unsigned short* __restrict__ keys,   // ks (sorted) or ksub
            const unsigned short* __restrict__ vt,     // vst [bh][d][N] or vsubt [bh][d][SS]
            const int* __restrict__ sblk,              // RES only
            const int* __restrict__ qsi,               // RES only (unsort scatter)
            float* __restrict__ attnb, float* __restrict__ lseb,
            float* __restrict__ out) {
  __shared__ __align__(16) unsigned short Pl[8][32][40];  // wave-private P scratch, padded
  const int w = threadIdx.x >> 5;
  const int lane = threadIdx.x & 31;
  const int lh = lane & 15;
  const int hs = lane >> 4;                 // lane-half select (K base 0 / 8)
  const int bh = blockIdx.x >> 5;
  const int qb = blockIdx.x & 31;
  const long long qbase = (long long)bh * NN + qb * BS + w * 32;

  // Q A-fragments (hoisted; constant across key chunks)
  FragU aq[2][2];
#pragma unroll
  for (int mt = 0; mt < 2; ++mt) {
    const unsigned short* rp = qs + (qbase + mt * 16 + lh) * DD;
#pragma unroll
    for (int kk = 0; kk < 2; ++kk) {
      int e = kk * 32 + hs * 8;
      aq[mt][kk].u[0] = *(const uint4*)(rp + e);
      aq[mt][kk].u[1] = *(const uint4*)(rp + e + 16);
    }
  }

  const unsigned short* kbase;
  const unsigned short* vbase;
  long long vstride;
  if (RES) { kbase = keys + (long long)bh * SS * DD;                 vbase = vt + (long long)bh * DD * SS;           vstride = SS; }
  else     { kbase = keys + ((long long)bh * NN + qb * BS) * DD;     vbase = vt + (long long)bh * DD * NN + qb * BS; vstride = NN; }

  const v8f vz = {0.f, 0.f, 0.f, 0.f, 0.f, 0.f, 0.f, 0.f};
  v8f O[2][4];
  float m_[2][8], l_[2][8];
#pragma unroll
  for (int mt = 0; mt < 2; ++mt) {
#pragma unroll
    for (int nt = 0; nt < 4; ++nt) O[mt][nt] = vz;
#pragma unroll
    for (int r = 0; r < 8; ++r) { m_[mt][r] = M_INIT; l_[mt][r] = 0.0f; }
  }

  for (int kc = 0; kc < 8; ++kc) {
    const int key0 = kc * 32;
    // K^T B-fragments
    FragU bk[2][2];
#pragma unroll
    for (int nt = 0; nt < 2; ++nt) {
      const unsigned short* rp = kbase + (long long)(key0 + nt * 16 + lh) * DD;
#pragma unroll
      for (int kk = 0; kk < 2; ++kk) {
        int e = kk * 32 + hs * 8;
        bk[nt][kk].u[0] = *(const uint4*)(rp + e);
        bk[nt][kk].u[1] = *(const uint4*)(rp + e + 16);
      }
    }
    // S = Q K^T
    v8f S[2][2];
#pragma unroll
    for (int mt = 0; mt < 2; ++mt)
#pragma unroll
      for (int nt = 0; nt < 2; ++nt) {
        v8f s = vz;
        s = __builtin_amdgcn_wmma_f32_16x16x32_bf16(false, aq[mt][0].v, false, bk[nt][0].v, (short)0, s, false, false);
        s = __builtin_amdgcn_wmma_f32_16x16x32_bf16(false, aq[mt][1].v, false, bk[nt][1].v, (short)0, s, false, false);
        S[mt][nt] = s;
      }
    // residual mask (per-lane key column)
    float msk0 = 0.0f, msk1 = 0.0f;
    if (RES) {
      msk0 = (sblk[bh * SS + key0 + lh]      == qb) ? MASK_NEG : 0.0f;
      msk1 = (sblk[bh * SS + key0 + 16 + lh] == qb) ? MASK_NEG : 0.0f;
    }
    // online softmax + stage P (bf16) to wave-private LDS
#pragma unroll
    for (int mt = 0; mt < 2; ++mt) {
#pragma unroll
      for (int r = 0; r < 8; ++r) {
        float a = S[mt][0][r] * SCALEF + msk0;
        float b = S[mt][1][r] * SCALEF + msk1;
        float cm = fmaxf(a, b);
        cm = fmaxf(cm, __shfl_xor(cm, 1, 32));
        cm = fmaxf(cm, __shfl_xor(cm, 2, 32));
        cm = fmaxf(cm, __shfl_xor(cm, 4, 32));
        cm = fmaxf(cm, __shfl_xor(cm, 8, 32));
        float nm = fmaxf(m_[mt][r], cm);
        float al = __expf(m_[mt][r] - nm);
        float p0 = __expf(a - nm);
        float p1 = __expf(b - nm);
        float rs = p0 + p1;
        rs += __shfl_xor(rs, 1, 32);
        rs += __shfl_xor(rs, 2, 32);
        rs += __shfl_xor(rs, 4, 32);
        rs += __shfl_xor(rs, 8, 32);
        l_[mt][r] = l_[mt][r] * al + rs;
        m_[mt][r] = nm;
#pragma unroll
        for (int nt4 = 0; nt4 < 4; ++nt4) O[mt][nt4][r] *= al;
        int rloc = mt * 16 + r + hs * 8;
        Pl[w][rloc][lh] = f2bf(p0);
        Pl[w][rloc][16 + lh] = f2bf(p1);
      }
    }
    // P A-fragments from LDS (in-order per wave, no barrier needed)
    FragU ap[2];
#pragma unroll
    for (int mt = 0; mt < 2; ++mt) {
      const unsigned short* pp = &Pl[w][mt * 16 + lh][0];
      ap[mt].u[0] = *(const uint4*)(pp + hs * 8);
      ap[mt].u[1] = *(const uint4*)(pp + 16 + hs * 8);
    }
    // V^T B-fragments (rows of transposed V are contiguous)
    FragU bv[4];
#pragma unroll
    for (int nt4 = 0; nt4 < 4; ++nt4) {
      const unsigned short* vp = vbase + (long long)(nt4 * 16 + lh) * vstride + key0;
      bv[nt4].u[0] = *(const uint4*)(vp + hs * 8);
      bv[nt4].u[1] = *(const uint4*)(vp + 16 + hs * 8);
    }
#pragma unroll
    for (int mt = 0; mt < 2; ++mt)
#pragma unroll
      for (int nt4 = 0; nt4 < 4; ++nt4)
        O[mt][nt4] = __builtin_amdgcn_wmma_f32_16x16x32_bf16(false, ap[mt].v, false, bv[nt4].v, (short)0, O[mt][nt4], false, false);
  }

  // finalize
#pragma unroll
  for (int mt = 0; mt < 2; ++mt) {
#pragma unroll
    for (int r = 0; r < 8; ++r) {
      int srow = qb * BS + w * 32 + mt * 16 + r + hs * 8;   // sorted row within (b,h)
      long long rowg = (long long)bh * NN + srow;
      float linv = 1.0f / l_[mt][r];
      float lse = m_[mt][r] + __logf(l_[mt][r]);
      if (!RES) {
#pragma unroll
        for (int nt4 = 0; nt4 < 4; ++nt4) {
          int d = nt4 * 16 + lh;
          attnb[rowg * DD + d] = O[mt][nt4][r] * linv;
        }
        if (lh == 0) lseb[rowg] = lse;
      } else {
        float l2 = lse + LOG32;
        float l1 = lseb[rowg];
        float c = 1.0f / (1.0f + __expf(l2 - l1));
        long long orow = (long long)bh * NN + qsi[rowg];    // unsort scatter
#pragma unroll
        for (int nt4 = 0; nt4 < 4; ++nt4) {
          int d = nt4 * 16 + lh;
          float a1 = attnb[rowg * DD + d];
          float a2 = O[mt][nt4][r] * linv;
          out[orow * DD + d] = c * a1 + (1.0f - c) * a2;
        }
      }
    }
  }
}

extern "C" void kernel_launch(void* const* d_in, const int* in_sizes, int n_in,
                              void* d_out, int out_size, void* d_ws, size_t ws_size,
                              hipStream_t stream) {
  const float* q       = (const float*)d_in[0];
  const float* k       = (const float*)d_in[1];
  const float* v       = (const float*)d_in[2];
  const float* proj    = (const float*)d_in[3];
  const int*   sampled = (const int*)d_in[4];
  float* out = (float*)d_out;

  char* ws = (char*)d_ws;
  size_t off = 0;
  auto alloc = [&](size_t bytes) -> void* {
    void* p = ws + off;
    off += (bytes + 255) & ~(size_t)255;
    return p;
  };
  unsigned int* qh   = (unsigned int*)alloc((size_t)BH * NN * 4);
  unsigned int* kh   = (unsigned int*)alloc((size_t)BH * NN * 4);
  int* qsi           = (int*)alloc((size_t)BH * NN * 4);
  int* ksi           = (int*)alloc((size_t)BH * NN * 4);
  unsigned short* qsb  = (unsigned short*)alloc((size_t)BH * NN * DD * 2);
  unsigned short* ksb  = (unsigned short*)alloc((size_t)BH * NN * DD * 2);
  unsigned short* vst  = (unsigned short*)alloc((size_t)BH * DD * NN * 2);
  unsigned short* ksub = (unsigned short*)alloc((size_t)BH * SS * DD * 2);
  unsigned short* vsubt= (unsigned short*)alloc((size_t)BH * DD * SS * 2);
  int* sblk          = (int*)alloc((size_t)BH * SS * 4);
  float* attnb       = (float*)alloc((size_t)BH * NN * DD * 4);
  float* lseb        = (float*)alloc((size_t)BH * NN * 4);

  hash_kernel<<<2 * BH * NN / 8, 256, 0, stream>>>(q, k, proj, qh, kh);
  sort_kernel<<<2 * BH, 64, 0, stream>>>(qh, kh, qsi, ksi);
  gather_qkv<<<BH * NN * 4 / 256, 256, 0, stream>>>(q, k, v, qsi, ksi, qsb, ksb, vst);
  gather_sub<<<BH * SS * 4 / 256, 256, 0, stream>>>(k, v, ksi, sampled, ksub, vsubt, sblk);
  attn_kernel<0><<<BH * NBLK, 256, 0, stream>>>(qsb, ksb, vst, nullptr, nullptr, attnb, lseb, nullptr);
  attn_kernel<1><<<BH * NBLK, 256, 0, stream>>>(qsb, ksub, vsubt, sblk, qsi, attnb, lseb, out);
}
